// TaskMoE_83305185673383
// MI455X (gfx1250) — matrix operand order repaired and checked by reading, not compiled, not run on hardware
//
#include <hip/hip_runtime.h>
#include <hip/hip_bf16.h>
#include <cstdint>

#define T_TOK 8192
#define H_DIM 1024
#define E_NUM 8
#define LDA   40   // LDS row stride in halves (32 data + 8 pad) -> conflict-free b128 frag reads

typedef __attribute__((ext_vector_type(16))) __bf16 v16bf;
typedef __attribute__((ext_vector_type(8)))  __bf16 v8bf;
typedef __attribute__((ext_vector_type(8)))  float  v8f;

__device__ __forceinline__ uint32_t pack_bf16(float a, float b) {
    union { __bf16 h[2]; uint32_t u; } u;
    u.h[0] = (__bf16)a; u.h[1] = (__bf16)b;
    return u.u;
}

// ---------------------------------------------------------------------------
// Phase 1: gating. One wave (32 lanes) per token: logits = x[t,:] @ wg[:,E],
// top-2 of logits (== top-2 of softmax), renormalized gates via logistic form.
// ---------------------------------------------------------------------------
__global__ __launch_bounds__(256) void moe_gate_kernel(
    const float* __restrict__ x, const float* __restrict__ wg,
    int* __restrict__ topIdx, float* __restrict__ gatesOut) {
    const int wave = threadIdx.x >> 5;
    const int lane = threadIdx.x & 31;
    const int t = blockIdx.x * 8 + wave;
    const float* xr = x + (size_t)t * H_DIM;

    float acc[E_NUM];
#pragma unroll
    for (int e = 0; e < E_NUM; ++e) acc[e] = 0.f;

    for (int i = 0; i < H_DIM / 32; ++i) {
        const int h = i * 32 + lane;
        const float xv = xr[h];
        const float4* wr = (const float4*)(wg + (size_t)h * E_NUM);
        const float4 w0 = wr[0], w1 = wr[1];
        acc[0] += xv * w0.x; acc[1] += xv * w0.y;
        acc[2] += xv * w0.z; acc[3] += xv * w0.w;
        acc[4] += xv * w1.x; acc[5] += xv * w1.y;
        acc[6] += xv * w1.z; acc[7] += xv * w1.w;
    }
#pragma unroll
    for (int e = 0; e < E_NUM; ++e) {
#pragma unroll
        for (int off = 16; off > 0; off >>= 1)
            acc[e] += __shfl_xor(acc[e], off, 32);
    }
    if (lane == 0) {
        int e0 = 0;
#pragma unroll
        for (int e = 1; e < E_NUM; ++e) if (acc[e] > acc[e0]) e0 = e;
        int e1 = -1; float best = -3.4e38f;
#pragma unroll
        for (int e = 0; e < E_NUM; ++e)
            if (e != e0 && acc[e] > best) { best = acc[e]; e1 = e; }
        const float l0 = acc[e0], l1 = acc[e1];
        // p0/(p0+p1) = 1/(1+exp(l1-l0))  (softmax denominator cancels)
        const float g0 = 1.f / (1.f + expf(l1 - l0));
        const float g1 = 1.f / (1.f + expf(l0 - l1));
        topIdx[2 * t]     = e0;  topIdx[2 * t + 1]   = e1;
        gatesOut[2 * t]   = g0;  gatesOut[2 * t + 1] = g1;
    }
}

// ---------------------------------------------------------------------------
// Phase 2: deterministic per-expert token lists (one block per expert,
// block-wide Hillis-Steele scan -> token-ordered compaction, no atomics).
// ---------------------------------------------------------------------------
__global__ __launch_bounds__(256) void moe_route_kernel(
    const int* __restrict__ topIdx, const float* __restrict__ gates,
    int* __restrict__ cnt, int* __restrict__ toks, float* __restrict__ gw) {
    const int e = blockIdx.x;
    const int tid = threadIdx.x;
    __shared__ int scan[256];
    int running = 0;
    for (int base = 0; base < T_TOK; base += 256) {
        const int t = base + tid;
        const int i0 = topIdx[2 * t], i1 = topIdx[2 * t + 1];
        int match = 0; float g = 0.f;
        if (i0 == e)      { match = 1; g = gates[2 * t]; }
        else if (i1 == e) { match = 1; g = gates[2 * t + 1]; }
        scan[tid] = match;
        __syncthreads();
        for (int off = 1; off < 256; off <<= 1) {
            const int v = (tid >= off) ? scan[tid - off] : 0;
            __syncthreads();
            scan[tid] += v;
            __syncthreads();
        }
        if (match) {
            const int pos = running + scan[tid] - 1;
            toks[e * T_TOK + pos] = t;
            gw[e * T_TOK + pos]   = g;
        }
        running += scan[255];
        __syncthreads();
    }
    if (tid == 0) cnt[e] = running;
}

// ---------------------------------------------------------------------------
// Phase 3: gathered expert GEMM. Block = 128 tokens x 128 outputs, 8 waves,
// each wave = 32x64 (2x4 wmma tiles). fp32 -> bf16 on the fly into LDS,
// v_wmma_f32_16x16x32_bf16 inner product over K=1024 in chunks of 32.
// Combine: out[tok] += gate * (acc + bias) via f32 global atomics
// (exactly 2 adds per element onto 0 -> commutative -> deterministic).
// ---------------------------------------------------------------------------
__global__ __launch_bounds__(256) void moe_expert_gemm(
    const float* __restrict__ x, const float* __restrict__ W,
    const float* __restrict__ bias, const int* __restrict__ cnt,
    const int* __restrict__ toks, const float* __restrict__ gw,
    float* __restrict__ out) {
    const int e = blockIdx.z;
    const int c = cnt[e];
    const int mbase = blockIdx.y * 128;
    if (mbase >= c) return;
    const int nbase = blockIdx.x * 128;

    __shared__ __bf16 sA[128 * LDA];
    __shared__ __bf16 sB[128 * LDA];
    __shared__ int    sTok[128];
    __shared__ float  sGate[128];

    const int tid = threadIdx.x;
    if (tid < 128) {
        const int idx = mbase + tid;
        if (idx < c) { sTok[tid] = toks[e * T_TOK + idx]; sGate[tid] = gw[e * T_TOK + idx]; }
        else         { sTok[tid] = 0;                     sGate[tid] = 0.f; }
    }
    __syncthreads();

    // staging: thread -> (row, 16-col half-chunk)
    const int lr = tid >> 1;
    const int lc = (tid & 1) * 16;
    const float* aBase = x + (size_t)sTok[lr] * H_DIM + lc;
    const float* bBase = W + ((size_t)e * H_DIM + nbase + lr) * H_DIM + lc;
    uint32_t* aDst = (uint32_t*)&sA[lr * LDA + lc];
    uint32_t* bDst = (uint32_t*)&sB[lr * LDA + lc];

    const int lane = tid & 31;
    const int wid  = tid >> 5;
    const int wm = wid >> 1;                 // 0..3  (rows)
    const int wn = wid & 1;                  // 0..1  (cols)
    const int aRow0 = wm * 32 + (lane & 15);
    const int aK0   = (lane >> 4) * 8;       // A: K split {0..7,16..23}|{8..15,24..31}
    const int bRow0 = wn * 64 + (lane & 15);
    const int bK0   = (lane >> 4) * 8;       // B: K split {0..7,16..23}|{8..15,24..31}

    v8f acc[2][4];
    {
        const v8f z = {};
#pragma unroll
        for (int i = 0; i < 2; ++i)
#pragma unroll
            for (int j = 0; j < 4; ++j) acc[i][j] = z;
    }

    for (int k0 = 0; k0 < H_DIM; k0 += 32) {
        const float4* ap = (const float4*)(aBase + k0);
        const float4* bp = (const float4*)(bBase + k0);
        const float4 a0 = ap[0], a1 = ap[1], a2 = ap[2], a3 = ap[3];
        const float4 b0 = bp[0], b1 = bp[1], b2 = bp[2], b3 = bp[3];
        if (k0 + 32 < H_DIM) {   // prefetch next K-chunk -> global_prefetch_b8
            __builtin_prefetch(aBase + k0 + 32, 0, 1);
            __builtin_prefetch(bBase + k0 + 32, 0, 1);
        }
        aDst[0] = pack_bf16(a0.x, a0.y); aDst[1] = pack_bf16(a0.z, a0.w);
        aDst[2] = pack_bf16(a1.x, a1.y); aDst[3] = pack_bf16(a1.z, a1.w);
        aDst[4] = pack_bf16(a2.x, a2.y); aDst[5] = pack_bf16(a2.z, a2.w);
        aDst[6] = pack_bf16(a3.x, a3.y); aDst[7] = pack_bf16(a3.z, a3.w);
        bDst[0] = pack_bf16(b0.x, b0.y); bDst[1] = pack_bf16(b0.z, b0.w);
        bDst[2] = pack_bf16(b1.x, b1.y); bDst[3] = pack_bf16(b1.z, b1.w);
        bDst[4] = pack_bf16(b2.x, b2.y); bDst[5] = pack_bf16(b2.z, b2.w);
        bDst[6] = pack_bf16(b3.x, b3.y); bDst[7] = pack_bf16(b3.z, b3.w);
        __syncthreads();

        v16bf aF[2], bF[4];
#pragma unroll
        for (int mi = 0; mi < 2; ++mi) {
            const int row = aRow0 + mi * 16;
            const v8bf lo = *(const v8bf*)&sA[row * LDA + aK0];
            const v8bf hi = *(const v8bf*)&sA[row * LDA + aK0 + 16];
            aF[mi] = __builtin_shufflevector(lo, hi, 0, 1, 2, 3, 4, 5, 6, 7,
                                             8, 9, 10, 11, 12, 13, 14, 15);
        }
#pragma unroll
        for (int ni = 0; ni < 4; ++ni) {
            const int row = bRow0 + ni * 16;
            const v8bf lo = *(const v8bf*)&sB[row * LDA + bK0];
            const v8bf hi = *(const v8bf*)&sB[row * LDA + bK0 + 16];
            bF[ni] = __builtin_shufflevector(lo, hi, 0, 1, 2, 3, 4, 5, 6, 7,
                                             8, 9, 10, 11, 12, 13, 14, 15);
        }
#pragma unroll
        for (int mi = 0; mi < 2; ++mi)
#pragma unroll
            for (int ni = 0; ni < 4; ++ni)
                acc[mi][ni] = __builtin_amdgcn_wmma_f32_16x16x32_bf16(
                    false, aF[mi], false, bF[ni], (short)0, acc[mi][ni],
                    false, false);
        __syncthreads();
    }

    // Epilogue: C/D layout -> element v of lane l is (M = v + (l>>4)*8, N = l&15)
    const int lrow8 = (lane >> 4) * 8;
    const int lcol  = lane & 15;
#pragma unroll
    for (int ni = 0; ni < 4; ++ni) {
        const int col = nbase + wn * 64 + ni * 16 + lcol;
        const float bc = bias[(size_t)e * H_DIM + col];
#pragma unroll
        for (int mi = 0; mi < 2; ++mi) {
#pragma unroll
            for (int v = 0; v < 8; ++v) {
                const int rl = wm * 32 + mi * 16 + lrow8 + v;
                if (mbase + rl < c) {
                    const float val = sGate[rl] * (acc[mi][ni][v] + bc);
                    unsafeAtomicAdd(out + (size_t)sTok[rl] * H_DIM + col, val);
                }
            }
        }
    }
}

// ---------------------------------------------------------------------------
extern "C" void kernel_launch(void* const* d_in, const int* in_sizes, int n_in,
                              void* d_out, int out_size, void* d_ws, size_t ws_size,
                              hipStream_t stream) {
    const float* x  = (const float*)d_in[0];   // [T, H]
    const float* wg = (const float*)d_in[1];   // [H, E]
    const float* W  = (const float*)d_in[2];   // [E, H, H]
    const float* b  = (const float*)d_in[3];   // [E, H]

    float* outMain = (float*)d_out;                                  // [T, H]
    int*   topIdx  = (int*)(outMain + (size_t)T_TOK * H_DIM);        // [T, 2] i32
    float* gatesO  = outMain + (size_t)T_TOK * H_DIM + 2 * T_TOK;    // [T, 2] f32

    int*   cnt  = (int*)d_ws;                                        // [E]
    int*   toks = (int*)((char*)d_ws + 256);                         // [E, T]
    float* gw   = (float*)((char*)d_ws + 256 +
                           sizeof(int) * (size_t)E_NUM * T_TOK);     // [E, T]

    hipMemsetAsync(d_out, 0, (size_t)T_TOK * H_DIM * sizeof(float), stream);
    moe_gate_kernel<<<T_TOK / 8, 256, 0, stream>>>(x, wg, topIdx, gatesO);
    moe_route_kernel<<<E_NUM, 256, 0, stream>>>(topIdx, gatesO, cnt, toks, gw);
    dim3 grid(H_DIM / 128, T_TOK / 128, E_NUM);
    moe_expert_gemm<<<grid, 256, 0, stream>>>(x, W, b, cnt, toks, gw, outMain);
}
